// GCN_encoder_22591527977028
// MI455X (gfx1250) — matrix operand order repaired and checked by reading, not compiled, not run on hardware
//
#include <hip/hip_runtime.h>
#include <hip/hip_bf16.h>
#include <stdint.h>

typedef __bf16 bf16x16 __attribute__((ext_vector_type(16)));
typedef float  f32x8   __attribute__((ext_vector_type(8)));
typedef unsigned int u32x4 __attribute__((ext_vector_type(4)));

#define N_GENES 978
#define BATCH   8192
#define KPAD    992      // 31 * 32
#define KSTEPS  (KPAD / 32)
#define HID     2048
#define NOUT    100
#define NOUT_PAD 112     // 7 * 16

__device__ __forceinline__ unsigned short f2bf(float f) {
  unsigned int u = __float_as_uint(f);
  u += 0x7FFFu + ((u >> 16) & 1u);          // round-to-nearest-even
  return (unsigned short)(u >> 16);
}

union BFrag { unsigned int u[8]; unsigned short h[16]; bf16x16 v; };
union Acc   { f32x8 v; float f[8]; };

__device__ __forceinline__ void ld_frag2(BFrag& fr, const unsigned short* p0,
                                         const unsigned short* p1) {
  u32x4 t0 = *reinterpret_cast<const u32x4*>(p0);
  u32x4 t1 = *reinterpret_cast<const u32x4*>(p1);
  fr.u[0] = t0[0]; fr.u[1] = t0[1]; fr.u[2] = t0[2]; fr.u[3] = t0[3];
  fr.u[4] = t1[0]; fr.u[5] = t1[1]; fr.u[6] = t1[2]; fr.u[7] = t1[3];
}

// ---------------- phase 0: zero the aggregation buffer ----------------
__global__ void k_zero(float* __restrict__ p, int n) {
  int i = blockIdx.x * blockDim.x + threadIdx.x;
  if (i < n) p[i] = 0.0f;
}

// ---------------- phase 1: edge scatter-add (GCN message passing) -----
// 2 edges per thread: one b128 load of the src pair, one of the dst pair.
__global__ void k_scatter(const float* __restrict__ x,
                          const long long* __restrict__ edges,
                          const float* __restrict__ gw,
                          float* __restrict__ agg, int E) {
  int e = (blockIdx.x * blockDim.x + threadIdx.x) * 2;
  if (e + 1 < E) {
    longlong2 s = *reinterpret_cast<const longlong2*>(&edges[e]);
    longlong2 d = *reinterpret_cast<const longlong2*>(&edges[(long long)E + e]);
    float w = gw[0];
    atomicAdd(&agg[d.x], x[s.x] * w);
    atomicAdd(&agg[d.y], x[s.y] * w);
  } else if (e < E) {
    long long s = edges[e], d = edges[(long long)E + e];
    atomicAdd(&agg[d], x[s] * gw[0]);
  }
}

// ---------------- phase 2: bias + ReLU + bf16 convert, K-padded -------
__global__ void k_bias_relu_bf16(const float* __restrict__ agg,
                                 const float* __restrict__ gb,
                                 unsigned short* __restrict__ hp, int total) {
  int i = blockIdx.x * blockDim.x + threadIdx.x;
  if (i >= total) return;
  int row = i / KPAD, col = i - row * KPAD;
  float v = 0.0f;
  if (col < N_GENES) {
    v = agg[row * N_GENES + col] + gb[0];
    v = v > 0.0f ? v : 0.0f;
  }
  hp[i] = f2bf(v);
}

// ------- weight converts: bf16, zero-padded, TRANSPOSED to [N][K] -----
__global__ void k_w1t(const float* __restrict__ W1, unsigned short* __restrict__ w1t, int total) {
  int i = blockIdx.x * blockDim.x + threadIdx.x;     // i = n*KPAD + k
  if (i >= total) return;
  int n = i / KPAD, k = i - n * KPAD;
  float v = (k < N_GENES) ? W1[k * HID + n] : 0.0f;
  w1t[i] = f2bf(v);
}
__global__ void k_w2t(const float* __restrict__ W2, unsigned short* __restrict__ w2t, int total) {
  int i = blockIdx.x * blockDim.x + threadIdx.x;     // i = n*HID + k
  if (i >= total) return;
  int n = i / HID, k = i - n * HID;
  float v = (n < NOUT) ? W2[k * NOUT + n] : 0.0f;
  w2t[i] = f2bf(v);
}

// ---------------- GEMM1: [8192 x 992]bf16 @ [992 x 2048]bf16 ----------
// block: 256 thr (8 waves), tile 128(M) x 128(N); wave = 64x32 (8 WMMA/K-step).
// Software-pipelined global->reg->LDS staging; fused bias+ReLU+bf16 store.
__global__ __launch_bounds__(256)
void k_gemm1(const unsigned short* __restrict__ A,   // hp  [8192][992]  ([m][k])
             const unsigned short* __restrict__ Bt,  // w1t [2048][992]  ([n][k])
             const float* __restrict__ bias,
             unsigned short* __restrict__ C) {       // h2  [8192][2048] bf16
  __shared__ __align__(16) unsigned short sA[128 * 32];  // [m][k]
  __shared__ __align__(16) unsigned short sB[128 * 32];  // [n][k]
  const int tid  = threadIdx.x;
  const int mBase = blockIdx.y * 128;
  const int nBase = blockIdx.x * 128;
  const int wave = tid >> 5, lane = tid & 31;
  const int half = lane >> 4, lmod = lane & 15;
  const int waveM = wave & 1, waveN = wave >> 1;      // 2 x 4 wave grid

  const unsigned int* gA = reinterpret_cast<const unsigned int*>(A);
  const unsigned int* gB = reinterpret_cast<const unsigned int*>(Bt);
  unsigned int* lA = reinterpret_cast<unsigned int*>(sA);
  unsigned int* lB = reinterpret_cast<unsigned int*>(sB);

  Acc acc[8];                                        // [mi(4)][nj(2)]
  #pragma unroll
  for (int i = 0; i < 8; ++i)
    #pragma unroll
    for (int j = 0; j < 8; ++j) acc[i].f[j] = 0.0f;

  unsigned int aReg[8], bReg[8];
  const int stRow = tid >> 4, stCol = tid & 15;      // staging: idx = tid + i*256

  // prologue: stage tile 0
  #pragma unroll
  for (int i = 0; i < 8; ++i) {
    int row = stRow + i * 16;
    aReg[i] = gA[((mBase + row) * KPAD) / 2 + stCol];
    bReg[i] = gB[((nBase + row) * KPAD) / 2 + stCol];
  }
  #pragma unroll
  for (int i = 0; i < 8; ++i) {
    lA[tid + i * 256] = aReg[i];
    lB[tid + i * 256] = bReg[i];
  }
  __syncthreads();

  for (int kb = 0; kb < KSTEPS; ++kb) {
    // issue next tile's global loads (overlap with LDS compute)
    if (kb + 1 < KSTEPS) {
      #pragma unroll
      for (int i = 0; i < 8; ++i) {
        int row = stRow + i * 16;
        aReg[i] = gA[((mBase + row) * KPAD + (kb + 1) * 32) / 2 + stCol];
        bReg[i] = gB[((nBase + row) * KPAD + (kb + 1) * 32) / 2 + stCol];
      }
    }
    // load all fragments into distinct registers, then WMMA back-to-back
    BFrag bf[2], af[4];
    #pragma unroll
    for (int j = 0; j < 2; ++j) {
      int n = waveN * 32 + j * 16 + lmod;
      ld_frag2(bf[j], &sB[n * 32 + half * 8], &sB[n * 32 + 16 + half * 8]);
    }
    #pragma unroll
    for (int i = 0; i < 4; ++i) {
      int row = waveM * 64 + i * 16 + lmod;
      ld_frag2(af[i], &sA[row * 32 + half * 8], &sA[row * 32 + 16 + half * 8]);
    }
    #pragma unroll
    for (int i = 0; i < 4; ++i)
      #pragma unroll
      for (int j = 0; j < 2; ++j)
        acc[i * 2 + j].v = __builtin_amdgcn_wmma_f32_16x16x32_bf16(
            false, af[i].v, false, bf[j].v, (short)0, acc[i * 2 + j].v, false, false);
    __syncthreads();
    if (kb + 1 < KSTEPS) {
      #pragma unroll
      for (int i = 0; i < 8; ++i) {
        lA[tid + i * 256] = aReg[i];
        lB[tid + i * 256] = bReg[i];
      }
    }
    __syncthreads();
  }

  // epilogue: bias + ReLU + bf16 store
  #pragma unroll
  for (int j = 0; j < 2; ++j) {
    int n = nBase + waveN * 32 + j * 16 + lmod;
    float bv = bias[n];
    #pragma unroll
    for (int i = 0; i < 4; ++i) {
      #pragma unroll
      for (int v = 0; v < 8; ++v) {
        int m = mBase + waveM * 64 + i * 16 + half * 8 + v;
        float val = acc[i * 2 + j].f[v] + bv;
        val = val > 0.0f ? val : 0.0f;
        C[m * HID + n] = f2bf(val);
      }
    }
  }
}

// ---------------- GEMM2: [8192 x 2048]bf16 @ [2048 x 112]bf16 ---------
// block: 128 thr (4 waves), tile 64(M) x 112(N); wave = 16 x 112 (7 WMMA/K-step)
__global__ __launch_bounds__(128)
void k_gemm2(const unsigned short* __restrict__ A,   // h2  [8192][2048] ([m][k])
             const unsigned short* __restrict__ Bt,  // w2t [112][2048]  ([n][k])
             const float* __restrict__ bias,
             float* __restrict__ out) {              // [8192][100] f32
  __shared__ __align__(16) unsigned short sA[64 * 32];   // [m][k]
  __shared__ __align__(16) unsigned short sB[112 * 32];  // [n][k]
  const int tid  = threadIdx.x;
  const int mBase = blockIdx.x * 64;
  const int wave = tid >> 5, lane = tid & 31;
  const int half = lane >> 4, lmod = lane & 15;

  const unsigned int* gA = reinterpret_cast<const unsigned int*>(A);
  const unsigned int* gB = reinterpret_cast<const unsigned int*>(Bt);
  unsigned int* lA = reinterpret_cast<unsigned int*>(sA);
  unsigned int* lB = reinterpret_cast<unsigned int*>(sB);

  Acc acc[7];
  #pragma unroll
  for (int g = 0; g < 7; ++g)
    #pragma unroll
    for (int j = 0; j < 8; ++j) acc[g].f[j] = 0.0f;

  const int stRow = tid >> 4, stCol = tid & 15;      // staging: idx = tid + i*128

  for (int kb = 0; kb < HID / 32; ++kb) {
    // stage A tile 64x32 (1024 dwords, 8/thread)
    #pragma unroll
    for (int i = 0; i < 8; ++i) {
      int row = stRow + i * 8;
      lA[tid + i * 128] = gA[((mBase + row) * HID + kb * 32) / 2 + stCol];
    }
    // stage B tile 112x32 (1792 dwords, 14/thread), already [n][k]
    #pragma unroll
    for (int i = 0; i < 14; ++i) {
      int row = stRow + i * 8;
      lB[tid + i * 128] = gB[(row * HID + kb * 32) / 2 + stCol];
    }
    __syncthreads();

    BFrag af, bf[7];
    {
      int row = wave * 16 + lmod;
      ld_frag2(af, &sA[row * 32 + half * 8], &sA[row * 32 + 16 + half * 8]);
    }
    #pragma unroll
    for (int g = 0; g < 7; ++g) {
      int n = g * 16 + lmod;
      ld_frag2(bf[g], &sB[n * 32 + half * 8], &sB[n * 32 + 16 + half * 8]);
    }
    #pragma unroll
    for (int g = 0; g < 7; ++g)
      acc[g].v = __builtin_amdgcn_wmma_f32_16x16x32_bf16(
          false, af.v, false, bf[g].v, (short)0, acc[g].v, false, false);
    __syncthreads();
  }

  // epilogue: bias add, f32 store, drop N padding
  #pragma unroll
  for (int g = 0; g < 7; ++g) {
    int n = g * 16 + lmod;
    if (n < NOUT) {
      float bv = bias[n];
      #pragma unroll
      for (int v = 0; v < 8; ++v) {
        int m = mBase + wave * 16 + half * 8 + v;
        out[m * NOUT + n] = acc[g].f[v] + bv;
      }
    }
  }
}

extern "C" void kernel_launch(void* const* d_in, const int* in_sizes, int n_in,
                              void* d_out, int out_size, void* d_ws, size_t ws_size,
                              hipStream_t stream) {
  const float*     inputs = (const float*)d_in[0];
  const long long* edges  = (const long long*)d_in[1];
  const float*     gw     = (const float*)d_in[2];
  const float*     gb     = (const float*)d_in[3];
  const float*     W1     = (const float*)d_in[4];
  const float*     b1     = (const float*)d_in[5];
  const float*     W2     = (const float*)d_in[6];
  const float*     b2     = (const float*)d_in[7];
  float* out = (float*)d_out;

  const int Nn = BATCH * N_GENES;       // 8,011,776 nodes
  const int E  = in_sizes[1] / 2;       // 20,000,000 edges

  // workspace carve-out
  char* ws = (char*)d_ws;
  size_t off = 0;
  auto carve = [&](size_t bytes) -> char* {
    char* p = ws + off;
    off += (bytes + 255) & ~(size_t)255;
    return p;
  };
  float*          agg = (float*)         carve((size_t)Nn * 4);             // 32.0 MB
  unsigned short* hp  = (unsigned short*)carve((size_t)BATCH * KPAD * 2);   // 16.3 MB
  unsigned short* w1t = (unsigned short*)carve((size_t)HID * KPAD * 2);     //  4.1 MB
  unsigned short* h2  = (unsigned short*)carve((size_t)BATCH * HID * 2);    // 33.6 MB
  unsigned short* w2t = (unsigned short*)carve((size_t)NOUT_PAD * HID * 2); //  0.5 MB

  k_zero<<<(Nn + 255) / 256, 256, 0, stream>>>(agg, Nn);
  k_w1t<<<(HID * KPAD + 255) / 256, 256, 0, stream>>>(W1, w1t, HID * KPAD);
  k_w2t<<<(NOUT_PAD * HID + 255) / 256, 256, 0, stream>>>(W2, w2t, NOUT_PAD * HID);
  k_scatter<<<(E / 2 + 255) / 256, 256, 0, stream>>>(inputs, edges, gw, agg, E);
  k_bias_relu_bf16<<<(BATCH * KPAD + 255) / 256, 256, 0, stream>>>(agg, gb, hp, BATCH * KPAD);
  k_gemm1<<<dim3(HID / 128, BATCH / 128), 256, 0, stream>>>(hp, w1t, b1, h2);
  k_gemm2<<<BATCH / 64, 128, 0, stream>>>(h2, w2t, b2, out);
}
